// SpikingMLP_29592324670184
// MI455X (gfx1250) — compile-verified, matrix operand order checked by reading, not run on hardware
//
#include <hip/hip_runtime.h>

// ---------------------------------------------------------------------------
// SpikingMLP on gfx1250: 25 steps x 3 layers of fused (bf16 WMMA GEMM + LIF),
// 4-stage async global->LDS pipeline (ASYNCcnt), ds_load_b128 fragment reads.
// ---------------------------------------------------------------------------

typedef __attribute__((ext_vector_type(16))) __bf16 v16bf;
typedef __attribute__((ext_vector_type(8)))  float  v8f;
typedef __attribute__((__vector_size__(4 * sizeof(int)))) int v4i;

#define T_STEPS 25
#define BATCH   512
#define D0 1024
#define D1 2048
#define D2 2048
#define D3 1024
#define BETA   0.95f
#define THRESH 1.0f
#define LDS_PITCH   80                    // 64B tile row + 16B pad (bank spread)
#define B_OFF       (64 * LDS_PITCH)      // B tile starts after 64-row A tile
#define STAGE_BYTES (192 * LDS_PITCH)     // A tile (64 rows) + B tile (128 rows)
#define NSTAGE 4                          // stage select = (i & 3)

// ---- CDNA5 async global->LDS + asynccnt helpers ----
__device__ __forceinline__ void async_copy_b128(const void* g, void* l) {
#if __has_builtin(__builtin_amdgcn_global_load_async_to_lds_b128)
  v4i* gg = (v4i*)const_cast<void*>(g);
  v4i* ll = (v4i*)l;
  __builtin_amdgcn_global_load_async_to_lds_b128(
      (__attribute__((address_space(1))) v4i*)gg,
      (__attribute__((address_space(3))) v4i*)ll, 0, 0);
#else
  asm volatile("global_load_async_to_lds_b128 %0, %1, off"
               :: "v"((unsigned)(unsigned long long)l), "v"(g) : "memory");
#endif
}
__device__ __forceinline__ void wait_async_le6() {
#if __has_builtin(__builtin_amdgcn_s_wait_asynccnt)
  __builtin_amdgcn_s_wait_asynccnt(6);
#else
  asm volatile("s_wait_asynccnt 0x6" ::: "memory");
#endif
}
__device__ __forceinline__ void wait_async_le0() {
#if __has_builtin(__builtin_amdgcn_s_wait_asynccnt)
  __builtin_amdgcn_s_wait_asynccnt(0);
#else
  asm volatile("s_wait_asynccnt 0x0" ::: "memory");
#endif
}

__device__ __forceinline__ unsigned short f32_to_bf16_rne(float f) {
  unsigned u = __builtin_bit_cast(unsigned, f);
  u += 0x7FFFu + ((u >> 16) & 1u);
  return (unsigned short)(u >> 16);
}

// x [B, T, D0] f32  ->  xbf [T, B, D0] bf16
__global__ __launch_bounds__(256) void cvt_x_kernel(const float* __restrict__ x,
                                                    unsigned short* __restrict__ xbf) {
  int i = blockIdx.x * 256 + threadIdx.x;
  int d = i & (D0 - 1);
  int b = (i >> 10) & (BATCH - 1);
  int t = i >> 19;
  xbf[i] = f32_to_bf16_rne(x[((size_t)b * T_STEPS + t) * D0 + d]);
}

// W [K, N] f32  ->  Wt [N, K] bf16
__global__ __launch_bounds__(256) void transpose_cvt_kernel(const float* __restrict__ W,
                                                            unsigned short* __restrict__ Wt,
                                                            int K, int N) {
  __shared__ float tile[32][33];
  int n0 = blockIdx.x * 32, k0 = blockIdx.y * 32;
  int tx = threadIdx.x, ty = threadIdx.y;   // blockDim = (32, 8)
#pragma unroll
  for (int r = 0; r < 32; r += 8)
    tile[ty + r][tx] = W[(size_t)(k0 + ty + r) * N + n0 + tx];
  __syncthreads();
#pragma unroll
  for (int r = 0; r < 32; r += 8)
    Wt[(size_t)(n0 + ty + r) * K + k0 + tx] = f32_to_bf16_rne(tile[tx][ty + r]);
}

union FragU { v16bf v; uint4 q[2]; };

// C = A[BATCH,K](bf16) x Wt[N,K]^T(bf16), fused LIF.
// Block tile 64(M) x 128(N), 128 threads = 4 waves (2x2), wave tile 32x64:
// per K-slab per wave: 6 fragments (12 ds_load_b128) -> 8 v_wmma_f32_16x16x32_bf16.
template <int K, int N, bool LAST>
__global__ __launch_bounds__(128) void lif_gemm_kernel(
    const unsigned short* __restrict__ A,    // [BATCH, K] bf16
    const unsigned short* __restrict__ Wt,   // [N, K] bf16
    float* __restrict__ mem,                 // [BATCH, N]
    unsigned short* __restrict__ spk_out,    // [BATCH, N] bf16 (when !LAST)
    float* __restrict__ out_spk,             // [BATCH, N] f32  (when LAST)
    float* __restrict__ out_mem)             // [BATCH, N] f32  (when LAST)
{
  __shared__ __attribute__((aligned(16))) char lds[NSTAGE * STAGE_BYTES];

  const int tid   = threadIdx.x;
  const int lane  = tid & 31;
  const int wave  = tid >> 5;      // 0..3
  const int waveM = wave >> 1;     // 0..1  (32 rows each)
  const int waveN = wave & 1;      // 0..1  (64 cols each)
  const int l15   = lane & 15;
  const int hi    = lane >> 4;

  const int mBase = blockIdx.y * 64;
  const int nBase = blockIdx.x * 128;
  const char* Ac = (const char*)A;
  const char* Wc = (const char*)Wt;

  v8f acc[2][4] = {};

  // Issue one K-slab (A: 64x32, B: 128x32 bf16): 6 async b128 loads / thread.
  auto issue = [&](int kb, int so) {
#pragma unroll
    for (int ii = 0; ii < 6; ++ii) {
      int c = tid + ii * 128;
      if (c < 256) {                     // A tile: 64 rows x 64B
        int row = c >> 2, off = (c & 3) * 16;
        async_copy_b128(Ac + ((size_t)(mBase + row) * K + kb) * 2 + off,
                        &lds[so + row * LDS_PITCH + off]);
      } else {                           // B tile: 128 rows x 64B
        int cb = c - 256, row = cb >> 2, off = (cb & 3) * 16;
        async_copy_b128(Wc + ((size_t)(nBase + row) * K + kb) * 2 + off,
                        &lds[so + B_OFF + row * LDS_PITCH + off]);
      }
    }
  };

  // Consume one staged K-slab: 6 ds_load_b128 fragments + 8 WMMAs.
  auto consume = [&](int so) {
    FragU af[2], bfr[4];
#pragma unroll
    for (int mf = 0; mf < 2; ++mf) {
      int row = waveM * 32 + mf * 16 + l15;
      int p = so + row * LDS_PITCH + hi * 16;
      af[mf].q[0] = *(const uint4*)&lds[p];
      af[mf].q[1] = *(const uint4*)&lds[p + 32];
    }
#pragma unroll
    for (int nf = 0; nf < 4; ++nf) {
      int row = waveN * 64 + nf * 16 + l15;
      int p = so + B_OFF + row * LDS_PITCH + hi * 16;
      bfr[nf].q[0] = *(const uint4*)&lds[p];
      bfr[nf].q[1] = *(const uint4*)&lds[p + 32];
    }
#pragma unroll
    for (int mf = 0; mf < 2; ++mf)
#pragma unroll
      for (int nf = 0; nf < 4; ++nf)
        acc[mf][nf] = __builtin_amdgcn_wmma_f32_16x16x32_bf16(
            false, af[mf].v, false, bfr[nf].v, (short)0, acc[mf][nf], false, false);
  };

  constexpr int iters = K >> 5;   // 32 or 64
  issue(0, 0);
  issue(32, STAGE_BYTES);

  int i = 0;
  for (; i < iters - 2; ++i) {
    wait_async_le6();            // slab i landed (slab i+1 may stay in flight)
    __syncthreads();             // visible to all; slab i-2 readers done
    issue((i + 2) * 32, ((i + 2) & 3) * STAGE_BYTES);
    consume((i & 3) * STAGE_BYTES);
  }
  wait_async_le6();  __syncthreads();  consume((i & 3) * STAGE_BYTES);  ++i;
  wait_async_le0();  __syncthreads();  consume((i & 3) * STAGE_BYTES);

  // ---- fused LIF epilogue (C/D layout: N = lane&15, M = vgpr + 8*hi) ----
#pragma unroll
  for (int mf = 0; mf < 2; ++mf) {
#pragma unroll
    for (int nf = 0; nf < 4; ++nf) {
      int col = nBase + waveN * 64 + nf * 16 + l15;
#pragma unroll
      for (int r = 0; r < 8; ++r) {
        int rowg = mBase + waveM * 32 + mf * 16 + r + 8 * hi;
        size_t idx = (size_t)rowg * N + col;
        float cur = acc[mf][nf][r];
        float mv  = mem[idx];
        float rst = (mv > THRESH) ? 1.0f : 0.0f;        // reset from PREVIOUS mem
        float m2  = __builtin_fmaf(BETA, mv, cur) * (1.0f - rst);
        mem[idx] = m2;
        float s = (m2 > THRESH) ? 1.0f : 0.0f;
        if constexpr (LAST) {
          out_spk[idx] = s;
          out_mem[idx] = m2;
        } else {
          spk_out[idx] = f32_to_bf16_rne(s);
        }
      }
    }
  }
}

extern "C" void kernel_launch(void* const* d_in, const int* in_sizes, int n_in,
                              void* d_out, int out_size, void* d_ws, size_t ws_size,
                              hipStream_t stream) {
  (void)in_sizes; (void)n_in; (void)out_size; (void)ws_size;
  const float* x  = (const float*)d_in[0];
  const float* W0 = (const float*)d_in[1];
  const float* W1 = (const float*)d_in[2];
  const float* W2 = (const float*)d_in[3];
  float* out = (float*)d_out;

  // --- carve workspace (~56 MB) ---
  char* ws = (char*)d_ws;
  size_t off = 0;
  auto alloc = [&](size_t bytes) -> char* {
    char* p = ws + off; off += (bytes + 255) & ~(size_t)255; return p;
  };
  unsigned short* xbf  = (unsigned short*)alloc((size_t)T_STEPS * BATCH * D0 * 2);
  unsigned short* w0t  = (unsigned short*)alloc((size_t)D1 * D0 * 2);
  unsigned short* w1t  = (unsigned short*)alloc((size_t)D2 * D1 * 2);
  unsigned short* w2t  = (unsigned short*)alloc((size_t)D3 * D2 * 2);
  unsigned short* spk0 = (unsigned short*)alloc((size_t)BATCH * D1 * 2);
  unsigned short* spk1 = (unsigned short*)alloc((size_t)BATCH * D2 * 2);
  float* mem0 = (float*)alloc((size_t)BATCH * D1 * 4);
  float* mem1 = (float*)alloc((size_t)BATCH * D2 * 4);   // contiguous with mem0
  float* mem2 = (float*)alloc((size_t)BATCH * D3 * 4);   // contiguous with mem1

  // --- prologue: precision / layout conversion + zero state ---
  cvt_x_kernel<<<(T_STEPS * BATCH * D0) / 256, 256, 0, stream>>>(x, xbf);
  transpose_cvt_kernel<<<dim3(D1/32, D0/32), dim3(32, 8), 0, stream>>>(W0, w0t, D0, D1);
  transpose_cvt_kernel<<<dim3(D2/32, D1/32), dim3(32, 8), 0, stream>>>(W1, w1t, D1, D2);
  transpose_cvt_kernel<<<dim3(D3/32, D2/32), dim3(32, 8), 0, stream>>>(W2, w2t, D2, D3);
  (void)hipMemsetAsync(mem0, 0, (size_t)BATCH * (D1 + D2 + D3) * sizeof(float), stream);

  // --- 25 steps x 3 dependent fused GEMM+LIF launches ---
  const size_t outPlane = (size_t)BATCH * D3;
  float* out_mem_base = out + (size_t)T_STEPS * outPlane;
  for (int t = 0; t < T_STEPS; ++t) {
    lif_gemm_kernel<D0, D1, false><<<dim3(D1/128, BATCH/64), 128, 0, stream>>>(
        xbf + (size_t)t * BATCH * D0, w0t, mem0, spk0, nullptr, nullptr);
    lif_gemm_kernel<D1, D2, false><<<dim3(D2/128, BATCH/64), 128, 0, stream>>>(
        spk0, w1t, mem1, spk1, nullptr, nullptr);
    lif_gemm_kernel<D2, D3, true><<<dim3(D3/128, BATCH/64), 128, 0, stream>>>(
        spk1, w2t, mem2, nullptr, out + t * outPlane, out_mem_base + t * outPlane);
  }
}
